// LSTMIsoformer_36764920054446
// MI455X (gfx1250) — matrix operand order repaired and checked by reading, not compiled, not run on hardware
//
#include <hip/hip_runtime.h>
#include <hip/hip_bf16.h>

typedef __bf16 bf16_t;
typedef __bf16 bf16x16 __attribute__((ext_vector_type(16)));
typedef float  f32x8   __attribute__((ext_vector_type(8)));
typedef float  f32x2   __attribute__((ext_vector_type(2)));
typedef unsigned int u32x4 __attribute__((ext_vector_type(4)));
typedef int          i32x8 __attribute__((ext_vector_type(8)));
typedef int          i32x4 __attribute__((ext_vector_type(4)));

#define BATCH 64
#define SEQ   256
#define HID_D 256
#define G4    1024      /* 4*H */
#define ISO_N 160000
#define NGENE 60000

// ---------------- workspace layout (bytes) ----------------
static constexpr size_t WS_WHH0   = 0;                          // 64nt*8kt*1024B  = 512KB (bf16 B-tiles)
static constexpr size_t WS_W1CAT  = 512ull << 10;               // 64nt*16kt*1024B = 1MB
static constexpr size_t WS_BIAS0  = 1536ull << 10;              // 4KB f32
static constexpr size_t WS_BIAS1  = 1540ull << 10;              // 4KB f32
static constexpr size_t WS_ROWMAX = 1544ull << 10;              // 64 u32
static constexpr size_t WS_HS0    = 2ull << 20;                 // [S][B][H] bf16 = 8MB
static constexpr size_t WS_HLAST  = 10ull << 20;                // [B][H] f32 = 64KB
static constexpr size_t WS_HID    = (10ull << 20) + (64ull << 10);
static constexpr size_t WS_DENOM  = (10ull << 20) + (128ull << 10); // [B][NGENE] f32 = 15.36MB

// ---------------- LDS layout for lstm kernel (bytes) ----------------
#define LDS_GATES  0            /* bf16 [1024][64] = 128K */
#define LDS_H      (128 << 10)  /* bf16 [64][256]  = 32K  */
#define LDS_C      (160 << 10)  /* f32  [64][256]  = 64K  */
#define LDS_XBUF   (224 << 10)  /* f32  [64]       = 256B */
#define LDS_XSTG   (225 << 10)  /* bf16 [2][64][256] = 2x32K (TDM double buffer) */
#define LDS_XSTG_SZ (32 << 10)
#define LDS_TOTAL  (289 << 10)

__device__ inline f32x8 zero8() {
    f32x8 z = {0.0f,0.0f,0.0f,0.0f,0.0f,0.0f,0.0f,0.0f};
    return z;
}
__device__ inline unsigned f2ord(float f) {
    unsigned u = __float_as_uint(f);
    return (u & 0x80000000u) ? ~u : (u | 0x80000000u);
}
__device__ inline float ord2f(unsigned u) {
    return (u & 0x80000000u) ? __uint_as_float(u & 0x7fffffffu) : __uint_as_float(~u);
}

// ---- TDM: async copy of one contiguous row (n8 x 8 bytes) global -> LDS ----
// D# per CDNA5 ISA: group0 = {count, lds_addr, global_addr, type=2};
// group1 = {data_size=8B, tensor_dim0=n8, tensor_dim1=1, tile_dim0=n8, tile_dim1=1, stride0=n8}.
// Tracked by TENSORcnt; issuing wave waits with s_wait_tensorcnt, then a workgroup
// barrier publishes the LDS contents to the other waves.
// This toolchain's builtin is the 6-arg clang-23 form:
//   (uint32x4 g0, int32x8 g1, int32x4 g2, int32x4 g3, int32x8 g4, i32 cpol)
__device__ inline void tdm_load_row(const void* gptr, unsigned lds_off, unsigned n8) {
    unsigned long long ga = (unsigned long long)(uintptr_t)gptr;
    u32x4 g0;
    g0[0] = 1u;                                            // count=1 (valid user D#)
    g0[1] = lds_off;                                       // lds_addr (bytes)
    g0[2] = (unsigned)(ga & 0xffffffffu);                  // global_addr[31:0]
    g0[3] = (unsigned)((ga >> 32) & 0x01ffffffu) | (2u << 30);  // addr[56:32] | type=2
    i32x8 g1;
    g1[0] = (int)(3u << 16);                               // wg_mask=0, data_size=3 (8B)
    g1[1] = (int)((n8 & 0xffffu) << 16);                   // tensor_dim0[15:0] @ bits63:48
    g1[2] = (int)(((n8 >> 16) & 0xffffu) | (1u << 16));    // tensor_dim0[31:16], tensor_dim1=1
    g1[3] = (int)((n8 & 0xffffu) << 16);                   // tile_dim0 @ bits127:112
    g1[4] = 1;                                             // tile_dim1=1, tile_dim2=0
    g1[5] = (int)n8;                                       // tensor_dim0_stride[31:0]
    g1[6] = 0;                                             // stride0 hi, stride1 lo
    g1[7] = 0;
    i32x4 g2 = {0, 0, 0, 0};
    i32x4 g3 = {0, 0, 0, 0};
    i32x8 g4 = {0, 0, 0, 0, 0, 0, 0, 0};
    __builtin_amdgcn_tensor_load_to_lds(g0, g1, g2, g3, g4, 0);
}

// =================== prep: pack weights into WMMA B-tile layout ===================
// B-tile as this code packs it (32x16 bf16, wave32; A uses the same k order): lane L holds column n = L%16;
// lanes 0-15 hold K=0..15 (elems 0..15), lanes 16-31 hold K=16..31.
// Tile storage: tiles ordered (nt*KT + kt); within tile: lane*16 + e bf16 elements.
__global__ void pack_whh0_kernel(const float* __restrict__ W, bf16_t* __restrict__ dst) {
    const int idx  = blockIdx.x * 256 + threadIdx.x;   // < 64*8*512
    const int tile = idx >> 9;
    const int r    = idx & 511;
    const int lane = r >> 4;
    const int e    = r & 15;
    const int nt   = tile >> 3;
    const int kt   = tile & 7;
    const int n = nt * 16 + (lane & 15);
    const int k = kt * 32 + ((lane >> 4) << 4) + e;
    dst[idx] = (bf16_t)W[n * HID_D + k];
}

// packed [Whh1 (kt 0..7) | Wih1 (kt 8..15)] -> one K=512 reduction
__global__ void pack_cat_kernel(const float* __restrict__ Whh1, const float* __restrict__ Wih1,
                                bf16_t* __restrict__ dst) {
    const int idx  = blockIdx.x * 256 + threadIdx.x;   // < 64*16*512
    const int tile = idx >> 9;
    const int r    = idx & 511;
    const int lane = r >> 4;
    const int e    = r & 15;
    const int nt   = tile >> 4;
    const int kt   = tile & 15;
    const int n = nt * 16 + (lane & 15);
    float v;
    if (kt < 8) {
        const int k = kt * 32 + ((lane >> 4) << 4) + e;
        v = Whh1[n * HID_D + k];
    } else {
        const int k = (kt - 8) * 32 + ((lane >> 4) << 4) + e;
        v = Wih1[n * HID_D + k];
    }
    dst[idx] = (bf16_t)v;
}

__global__ void prep_misc_kernel(const float* __restrict__ bih0, const float* __restrict__ bhh0,
                                 const float* __restrict__ bih1, const float* __restrict__ bhh1,
                                 float* __restrict__ bias0, float* __restrict__ bias1,
                                 unsigned* __restrict__ rowmax, float* __restrict__ denom) {
    const int idx = blockIdx.x * 256 + threadIdx.x;
    if (idx < BATCH * NGENE) denom[idx] = 0.0f;
    if (idx < BATCH) rowmax[idx] = 0u;   // sortable encoding of very-negative
    if (idx < G4) {
        bias0[idx] = bih0[idx] + bhh0[idx];
        bias1[idx] = bih1[idx] + bhh1[idx];
    }
}

// =================== persistent LSTM layer (1 workgroup, 32 waves) ===================
// C-tile grid: M=64 (4 mt) x N=1024 (64 nt). Wave w owns nt = {2w, 2w+1}, processed
// one nt at a time (acc 32 + A 32 + B 8 VGPRs live -> no spills in the serial loop).
// Layer 1 input rows hs0[t] are staged into LDS by the Tensor Data Mover one step
// ahead (double buffered), so 32 waves read A-fragments from LDS instead of each
// re-streaming the same 32KB from global.
__global__ __launch_bounds__(1024)
void lstm_layer_kernel(const bf16_t* __restrict__ Bpk,    // packed B tiles
                       const float*  __restrict__ bias,   // fused bias [1024]
                       const float*  __restrict__ x,      // [B,S]   (layer0)
                       const float*  __restrict__ wih0,   // [1024]  (layer0)
                       const bf16_t* __restrict__ xs_in,  // [S,B,H] (layer1 input)
                       bf16_t* __restrict__ hs_out,       // [S,B,H] (layer0 output)
                       float*  __restrict__ hlast_out,    // [B,H]   (layer1 output)
                       int KT, int layer) {
    extern __shared__ char smem[];
    bf16_t* gatesS = (bf16_t*)(smem + LDS_GATES);
    bf16_t* hS     = (bf16_t*)(smem + LDS_H);
    float*  cS     = (float*) (smem + LDS_C);
    float*  xbuf   = (float*) (smem + LDS_XBUF);

    const int tid   = threadIdx.x;
    const int lane  = tid & 31;
    const int wave  = tid >> 5;
    const int l16   = lane & 15;
    const int lhalf = (lane >> 4) & 1;

    for (int i = tid; i < BATCH * HID_D; i += 1024) { hS[i] = (bf16_t)0.0f; cS[i] = 0.0f; }
    if (layer == 0 && tid < BATCH) xbuf[tid] = x[tid * SEQ];   // x[:, 0]
    if (layer == 1 && tid == 0) {                              // stage hs0[0] via TDM
        tdm_load_row(xs_in, LDS_XSTG, (BATCH * HID_D * 2) / 8);
        __builtin_amdgcn_s_wait_tensorcnt(0);
    }
    __syncthreads();

    // t-invariant per-nt scalars, hoisted out of the 256-step recurrence
    float bj[2], wj[2];
    int   jn[2];
#pragma unroll
    for (int nn = 0; nn < 2; ++nn) {
        jn[nn] = (wave * 2 + nn) * 16 + l16;
        bj[nn] = bias[jn[nn]];
        wj[nn] = (layer == 0) ? wih0[jn[nn]] : 0.0f;
    }
    // t-invariant A-fragment element offsets: (m row)*H + lane-half K offset
    int aoff[4];
#pragma unroll
    for (int mt = 0; mt < 4; ++mt) aoff[mt] = (mt * 16 + l16) * HID_D + lhalf * 16;

    for (int t = 0; t < SEQ; ++t) {
        // kick off async TDM stage of hs0[t+1] while step t computes
        if (layer == 1 && tid == 0 && t + 1 < SEQ)
            tdm_load_row(xs_in + (size_t)(t + 1) * BATCH * HID_D,
                         LDS_XSTG + ((t + 1) & 1) * LDS_XSTG_SZ,
                         (BATCH * HID_D * 2) / 8);
        const bf16_t* xst = (const bf16_t*)(smem + LDS_XSTG + (t & 1) * LDS_XSTG_SZ);

#pragma unroll
        for (int nn = 0; nn < 2; ++nn) {
            const int nt = wave * 2 + nn;
            f32x8 acc[4];
#pragma unroll
            for (int mt = 0; mt < 4; ++mt) acc[mt] = zero8();

            const bf16_t* bp = Bpk + (size_t)nt * KT * 512 + lane * 16;

            // ---- K-range 0..255: A = h (LDS) ----
            for (int kt = 0; kt < 8; ++kt) {
                union { bf16x16 v; uint4 q[2]; } a[4];
#pragma unroll
                for (int mt = 0; mt < 4; ++mt) {
                    const bf16_t* p = hS + aoff[mt] + kt * 32;
                    a[mt].q[0] = *(const uint4*)(p);
                    a[mt].q[1] = *(const uint4*)(p + 8);
                }
                union { bf16x16 v; uint4 q[2]; } bm;
                bm.q[0] = *(const uint4*)(bp);
                bm.q[1] = *(const uint4*)(bp + 8);
                bp += 512;
#pragma unroll
                for (int mt = 0; mt < 4; ++mt)
                    acc[mt] = __builtin_amdgcn_wmma_f32_16x16x32_bf16(
                        false, a[mt].v, false, bm.v, (short)0, acc[mt], false, false);
            }
            // ---- K-range 256..511 (layer1 only): A = hs0[t] (TDM-staged LDS) ----
            for (int kt = 8; kt < KT; ++kt) {
                union { bf16x16 v; uint4 q[2]; } a[4];
#pragma unroll
                for (int mt = 0; mt < 4; ++mt) {
                    const bf16_t* p = xst + aoff[mt] + (kt - 8) * 32;
                    a[mt].q[0] = *(const uint4*)(p);
                    a[mt].q[1] = *(const uint4*)(p + 8);
                }
                union { bf16x16 v; uint4 q[2]; } bm;
                bm.q[0] = *(const uint4*)(bp);
                bm.q[1] = *(const uint4*)(bp + 8);
                bp += 512;
#pragma unroll
                for (int mt = 0; mt < 4; ++mt)
                    acc[mt] = __builtin_amdgcn_wmma_f32_16x16x32_bf16(
                        false, a[mt].v, false, bm.v, (short)0, acc[mt], false, false);
            }

            // epilogue: + bias (+ x_t*Wih0 for layer0), activate, store bf16 gates to LDS
            // C layout: lane L, vgpr v -> m = v + 16*mt + (L>=16?8:0), n = L%16
            const int j    = jn[nn];          // gate index (whole tile in one gate)
            const int gate = j >> 8;          // 0:i 1:f 2:g 3:o
#pragma unroll
            for (int mt = 0; mt < 4; ++mt) {
                union { bf16_t h8[8]; uint4 q; } st;
#pragma unroll
                for (int v = 0; v < 8; ++v) {
                    const int bb = mt * 16 + lhalf * 8 + v;
                    float g = acc[mt][v] + bj[nn];
                    if (layer == 0) g += xbuf[bb] * wj[nn];
                    const float act = (gate == 2) ? tanhf(g) : (1.0f / (1.0f + __expf(-g)));
                    st.h8[v] = (bf16_t)act;
                }
                *(uint4*)(gatesS + j * BATCH + mt * 16 + lhalf * 8) = st.q;  // [j][b], 16B store
            }
        }
        __syncthreads();

        // elementwise cell/hidden update; also stage x[:, t+1] for next step
        for (int i = tid; i < BATCH * HID_D; i += 1024) {
            const int bb = i >> 8;
            const int u  = i & 255;
            const float ig = (float)gatesS[(u      ) * BATCH + bb];
            const float fg = (float)gatesS[(u + 256) * BATCH + bb];
            const float gg = (float)gatesS[(u + 512) * BATCH + bb];
            const float og = (float)gatesS[(u + 768) * BATCH + bb];
            const float c  = fg * cS[i] + ig * gg;
            cS[i] = c;
            const float h = og * tanhf(c);
            hS[i] = (bf16_t)h;
            if (layer == 0) hs_out[(size_t)t * BATCH * HID_D + i] = (bf16_t)h;
            if (layer == 1 && t == SEQ - 1) hlast_out[i] = h;
        }
        if (layer == 0 && tid < BATCH && t + 1 < SEQ) xbuf[tid] = x[tid * SEQ + t + 1];
        // publish next step's TDM-staged buffer before anyone reads it
        if (layer == 1 && tid == 0) __builtin_amdgcn_s_wait_tensorcnt(0);
        __syncthreads();
    }
}

// =================== fc1: relu(h_last @ W1^T + b1), tiny -> plain FMA ===================
__global__ void fc1_kernel(const float* __restrict__ hlast, const float* __restrict__ W1,
                           const float* __restrict__ b1, float* __restrict__ hid) {
    const int j = blockIdx.x * 256 + threadIdx.x;   // 64*256
    const int b = j >> 8, u = j & 255;
    float s = b1[u];
    const float* hr = hlast + b * HID_D;
    const float* wr = W1 + u * HID_D;
    for (int k = 0; k < HID_D; ++k) s += hr[k] * wr[k];
    hid[j] = fmaxf(s, 0.0f);
}

// =================== fc2: logits = hid @ W2^T + b2, fp32 WMMA, memory bound ===================
// W2 (164MB) is single-use: stream it with non-temporal loads so it doesn't
// evict the reused logits/hid working set from L2.
// f32 16x16x4: A lane L -> m=L%16, V0/V1 = K (L<16 ? 0,1 : 2,3); B lane L -> n=L%16, same K split.
__global__ __launch_bounds__(256)
void fc2_kernel(const float* __restrict__ hid, const float* __restrict__ W2,
                const float* __restrict__ b2, float* __restrict__ logits) {
    const int lane = threadIdx.x & 31;
    const int wave = threadIdx.x >> 5;
    const int nt   = blockIdx.x * 8 + wave;        // 0..9999
    const int l16  = lane & 15;
    const int koff = ((lane >> 4) & 1) * 2;

    f32x8 acc[4];
#pragma unroll
    for (int mt = 0; mt < 4; ++mt) acc[mt] = zero8();

    const float* w2row = W2 + (size_t)(nt * 16 + l16) * HID_D;
#pragma unroll 4
    for (int kt = 0; kt < 64; ++kt) {
        const int k0 = kt * 4 + koff;
        if ((kt & 15) == 0) __builtin_prefetch(w2row + kt * 4 + 256, 0, 0);
        const f32x2 bt = __builtin_nontemporal_load((const f32x2*)(w2row + k0));
#pragma unroll
        for (int mt = 0; mt < 4; ++mt) {
            const f32x2 at = *(const f32x2*)(hid + (mt * 16 + l16) * HID_D + k0);
            acc[mt] = __builtin_amdgcn_wmma_f32_16x16x4_f32(
                false, at, false, bt, (short)0, acc[mt], false, false);
        }
    }
    const int n = nt * 16 + l16;
    const float bn = b2[n];
    const int moff = ((lane >> 4) & 1) * 8;
#pragma unroll
    for (int mt = 0; mt < 4; ++mt)
#pragma unroll
        for (int v = 0; v < 8; ++v)
            logits[(size_t)(mt * 16 + moff + v) * ISO_N + n] = acc[mt][v] + bn;
}

// =================== grouped softmax ===================
__global__ void rowmax_kernel(const float* __restrict__ logits, unsigned* __restrict__ rowmax) {
    __shared__ float red[256];
    const int b = blockIdx.x >> 5;
    const int chunk = blockIdx.x & 31;
    const int base = chunk * (ISO_N / 32);   // 5000
    float m = -3.4e38f;
    for (int i = threadIdx.x; i < ISO_N / 32; i += 256)
        m = fmaxf(m, logits[(size_t)b * ISO_N + base + i]);
    red[threadIdx.x] = m;
    __syncthreads();
    for (int s = 128; s > 0; s >>= 1) {
        if (threadIdx.x < s) red[threadIdx.x] = fmaxf(red[threadIdx.x], red[threadIdx.x + s]);
        __syncthreads();
    }
    if (threadIdx.x == 0) atomicMax(rowmax + b, f2ord(red[0]));
}

__global__ void exp_scatter_kernel(float* __restrict__ ex, const unsigned* __restrict__ rowmax,
                                   const int* __restrict__ gidx, float* __restrict__ denom) {
    const int i = blockIdx.x * 256 + threadIdx.x;   // 625*256 = 160000
    const int b = blockIdx.y;
    const float m = ord2f(rowmax[b]);
    const size_t idx = (size_t)b * ISO_N + i;
    const float e = __expf(ex[idx] - m);
    ex[idx] = e;
    atomicAdd(denom + b * NGENE + gidx[i], e);
}

__global__ void norm_kernel(float* __restrict__ ex, const int* __restrict__ gidx,
                            const float* __restrict__ denom) {
    const int i = blockIdx.x * 256 + threadIdx.x;
    const int b = blockIdx.y;
    const size_t idx = (size_t)b * ISO_N + i;
    ex[idx] = ex[idx] / fmaxf(denom[b * NGENE + gidx[i]], 1e-8f);
}

// =================== launch ===================
extern "C" void kernel_launch(void* const* d_in, const int* in_sizes, int n_in,
                              void* d_out, int out_size, void* d_ws, size_t ws_size,
                              hipStream_t stream) {
    (void)in_sizes; (void)n_in; (void)out_size; (void)ws_size;
    const float* x    = (const float*)d_in[0];
    const float* Wih0 = (const float*)d_in[1];
    const float* Whh0 = (const float*)d_in[2];
    const float* bih0 = (const float*)d_in[3];
    const float* bhh0 = (const float*)d_in[4];
    const float* Wih1 = (const float*)d_in[5];
    const float* Whh1 = (const float*)d_in[6];
    const float* bih1 = (const float*)d_in[7];
    const float* bhh1 = (const float*)d_in[8];
    const float* W1   = (const float*)d_in[9];
    const float* b1   = (const float*)d_in[10];
    const float* W2   = (const float*)d_in[11];
    const float* b2   = (const float*)d_in[12];
    const int*   gidx = (const int*)d_in[13];

    char* ws = (char*)d_ws;
    bf16_t*   whh0_pk  = (bf16_t*)(ws + WS_WHH0);
    bf16_t*   w1cat_pk = (bf16_t*)(ws + WS_W1CAT);
    float*    bias0    = (float*)(ws + WS_BIAS0);
    float*    bias1    = (float*)(ws + WS_BIAS1);
    unsigned* rowmax   = (unsigned*)(ws + WS_ROWMAX);
    bf16_t*   hs0      = (bf16_t*)(ws + WS_HS0);
    float*    hlast    = (float*)(ws + WS_HLAST);
    float*    hid      = (float*)(ws + WS_HID);
    float*    denom    = (float*)(ws + WS_DENOM);
    float*    logits   = (float*)d_out;

    pack_whh0_kernel<<<(64 * 8 * 512) / 256, 256, 0, stream>>>(Whh0, whh0_pk);
    pack_cat_kernel<<<(64 * 16 * 512) / 256, 256, 0, stream>>>(Whh1, Wih1, w1cat_pk);
    prep_misc_kernel<<<(BATCH * NGENE + 255) / 256, 256, 0, stream>>>(
        bih0, bhh0, bih1, bhh1, bias0, bias1, rowmax, denom);

    lstm_layer_kernel<<<1, 1024, LDS_TOTAL, stream>>>(
        whh0_pk, bias0, x, Wih0, hs0, hs0, hlast, 8, 0);
    lstm_layer_kernel<<<1, 1024, LDS_TOTAL, stream>>>(
        w1cat_pk, bias1, x, Wih0, hs0, hs0, hlast, 16, 1);

    fc1_kernel<<<64, 256, 0, stream>>>(hlast, W1, b1, hid);
    fc2_kernel<<<1250, 256, 0, stream>>>(hid, W2, b2, logits);

    rowmax_kernel<<<64 * 32, 256, 0, stream>>>(logits, rowmax);
    exp_scatter_kernel<<<dim3(ISO_N / 256, BATCH), 256, 0, stream>>>(logits, rowmax, gidx, denom);
    norm_kernel<<<dim3(ISO_N / 256, BATCH), 256, 0, stream>>>(logits, gidx, denom);
}
